// VectorQuantizer_72825465471171
// MI455X (gfx1250) — matrix operand order repaired and checked by reading, not compiled, not run on hardware
//
#include <hip/hip_runtime.h>
#include <hip/hip_bf16.h>

typedef __attribute__((ext_vector_type(16))) _Float16     v16h;
typedef __attribute__((ext_vector_type(8)))  _Float16     v8h;
typedef __attribute__((ext_vector_type(8)))  float        v8f;
typedef __attribute__((ext_vector_type(4)))  unsigned int v4u;
typedef __attribute__((ext_vector_type(8)))  int          v8i;
typedef __attribute__((ext_vector_type(4)))  int          v4i;

namespace {
constexpr int    kNumEmb = 512;
constexpr int    kDim    = 64;
constexpr int    kHW     = 4096;     // 64*64
constexpr int    kCHW    = 262144;   // 64*4096
constexpr int    kNPix   = 131072;   // 32*64*64
constexpr int    kPitchH = 72;       // padded LDS row pitch in halves (36 dwords)
constexpr size_t kLossOff  = 0;
constexpr size_t kQuantOff = 131072;            // after loss [B,H,W]
constexpr size_t kPerpOff  = 131072 + 8388608;  // 8519680
constexpr size_t kEncOff   = kPerpOff + 1;      // 8519681 (odd -> b32 stores)
}

// ---------------------------------------------------------------------------
// Kernel 1: codebook prep — f32 emb -> f16 copy (for WMMA), ||e||^2, zero hist
// ---------------------------------------------------------------------------
__global__ __launch_bounds__(512) void vq_prep(const float* __restrict__ emb,
                                               _Float16* __restrict__ embh,
                                               float* __restrict__ enorm,
                                               unsigned* __restrict__ counts) {
  const int k = blockIdx.x * blockDim.x + threadIdx.x;
  if (k >= kNumEmb) return;
  float s = 0.f;
#pragma unroll 8
  for (int d = 0; d < kDim; ++d) {
    const float v = emb[k * kDim + d];
    embh[k * kDim + d] = (_Float16)v;
    s += v * v;
  }
  enorm[k] = s;
  counts[k] = 0u;
}

// ---------------------------------------------------------------------------
// Kernel 2: main VQ — one wave32 per 16-pixel tile, WMMA f32_16x16x32_f16.
// Codebook staged into LDS once per block via the Tensor Data Mover with
// LDS padding (36-dword row pitch) so ds_load_b128 of B fragments is
// bank-conflict free across the 16 columns of a half-wave.
// ---------------------------------------------------------------------------
__global__ __launch_bounds__(256) void vq_main(const float* __restrict__ x,
                                               const float* __restrict__ emb,
                                               const _Float16* __restrict__ embh,
                                               const float* __restrict__ enorm,
                                               unsigned* __restrict__ counts,
                                               float* __restrict__ out) {
  __shared__ __align__(128) _Float16 Bsh[kNumEmb * kPitchH];  // 72 KB

  const int lane = threadIdx.x & 31;
  const int tile = blockIdx.x * (blockDim.x >> 5) + (threadIdx.x >> 5);
  const int base = tile << 4;        // first pixel row of this 16-row tile
  const int nl   = lane & 15;
  const int h    = lane >> 4;        // half-wave select

  // ---- stage codebook (64 KB, padded to 72 KB) into LDS ----
#if __has_builtin(__builtin_amdgcn_tensor_load_to_lds)
  if (threadIdx.x < 32) {            // wave 0 drives the TDM (EXEC ignored)
    const unsigned long long ga = (unsigned long long)embh;
    v4u g0;
    g0[0] = 1u;                                         // count=1, user D#
    g0[1] = 0u;                                         // lds_addr = 0 (Bsh)
    g0[2] = (unsigned)(ga & 0xFFFFFFFFull);             // global_addr[31:0]
    g0[3] = (unsigned)((ga >> 32) & 0x01FFFFFFull)      // global_addr[56:32]
          | (2u << 30);                                 // type = 2 ("image")
    v8i g1;
    g1[0] = (int)((1u << 16)        // data_size = 1 -> 2-byte elements
          |       (1u << 20)        // pad_enable
          |       (4u << 22)        // pad_interval code 4 -> every 32 dwords
          |       (3u << 25));      // pad_amount  code 3 -> 4 dwords
    g1[1] = (int)(64u  << 16);      // tensor_dim0 = 64   (bits 79:48 lo)
    g1[2] = (int)(512u << 16);      // tensor_dim1 = 512  (bits 111:80 lo)
    g1[3] = (int)(64u  << 16);      // tile_dim0  = 64    (bits 127:112)
    g1[4] = (int)512;               // tile_dim1 = 512, tile_dim2 = 0
    g1[5] = 64;                     // tensor_dim0_stride = 64
    g1[6] = 0;                      // stride hi / tensor_dim1_stride lo
    g1[7] = 0;
    const v4i z4 = {0, 0, 0, 0};
#if __clang_major__ >= 23
    const v8i z8 = {0, 0, 0, 0, 0, 0, 0, 0};
    __builtin_amdgcn_tensor_load_to_lds(g0, g1, z4, z4, z8, 0);
#else
    __builtin_amdgcn_tensor_load_to_lds(g0, g1, z4, z4, 0);
#endif
    __builtin_amdgcn_s_wait_tensorcnt(0);
  }
  // The TDM write to LDS is invisible to alias analysis (it only sees the
  // descriptor words): escape Bsh into an opaque asm with a memory clobber so
  // subsequent ds loads are not folded to undef.
  {
    _Float16* p = &Bsh[0];
    asm volatile("" : "+v"(p) : : "memory");
  }
  __syncthreads();
#else
  // fallback: cooperative vector copy into the same padded layout
  for (int c = threadIdx.x; c < kNumEmb * 4; c += blockDim.x) {
    const int code = c >> 2, seg = c & 3;               // 16-half segments
    *(v8h*)(Bsh + code * kPitchH + seg * 16)     = *(const v8h*)(embh + code * kDim + seg * 16);
    *(v8h*)(Bsh + code * kPitchH + seg * 16 + 8) = *(const v8h*)(embh + code * kDim + seg * 16 + 8);
  }
  __syncthreads();
#endif

  __builtin_prefetch(enorm + nl, 0, 1);

  // ---- A tile: 16 rows x 64 dims; NCHW gather (stride HW) -> f16 ----
  // 16-bit A 16x32 layout: lane (h,nl): elems 0..7 -> K=8h+j, elems 8..15 -> K=16+8h+j
  v16h a0, a1;
  {
    const int n  = base + nl;
    const int bb = n >> 12;
    const int hw = n & (kHW - 1);
    const float* xr = x + (size_t)bb * kCHW + hw;
#pragma unroll
    for (int j = 0; j < 8; ++j) {
      a0[j]     = (_Float16)xr[(size_t)(8 * h + j) * kHW];
      a0[j + 8] = (_Float16)xr[(size_t)(16 + 8 * h + j) * kHW];
      a1[j]     = (_Float16)xr[(size_t)(32 + 8 * h + j) * kHW];
      a1[j + 8] = (_Float16)xr[(size_t)(48 + 8 * h + j) * kHW];
    }
  }

  // running argmin of (||e||^2 - 2 x·e); ||x||^2 constant per row -> dropped
  float bestv[8];
  int   besti[8];
#pragma unroll
  for (int v = 0; v < 8; ++v) { bestv[v] = 3.4e38f; besti[v] = 0; }

  for (int ct = 0; ct < kNumEmb / 16; ++ct) {
    const int code = ct * 16 + nl;
    const _Float16* brow = Bsh + (size_t)code * kPitchH;
    // 16-bit B 32x16 layout: lane (h,nl)=col nl, elems 0..7 -> K=8h+j, elems 8..15 -> K=16+8h+j
    // padded rows are 16B-aligned: assemble v16h from two v8h ds_load_b128
    const v8h* p0 = (const v8h*)(brow + 8 * h);         // K-tile 0: d 0..31
    const v8h* p1 = (const v8h*)(brow + 32 + 8 * h);    // K-tile 1: d 32..63
    v16h b0, b1;
#pragma unroll
    for (int j = 0; j < 8; ++j) {
      b0[j] = p0[0][j]; b0[j + 8] = p0[2][j];
      b1[j] = p1[0][j]; b1[j + 8] = p1[2][j];
    }
    v8f acc = {};
    acc = __builtin_amdgcn_wmma_f32_16x16x32_f16(false, a0, false, b0,
                                                 (short)0, acc, false, false);
    acc = __builtin_amdgcn_wmma_f32_16x16x32_f16(false, a1, false, b1,
                                                 (short)0, acc, false, false);
    const float en = enorm[code];
#pragma unroll
    for (int v = 0; v < 8; ++v) {          // acc[v]: row M=v+8h, col N=nl
      const float cand = en - 2.0f * acc[v];
      if (cand < bestv[v]) { bestv[v] = cand; besti[v] = code; }
    }
  }

  // min-reduce across the 16 columns held within each half-wave
#pragma unroll
  for (int v = 0; v < 8; ++v) {
#pragma unroll
    for (int off = 8; off >= 1; off >>= 1) {
      const float ov = __shfl_xor(bestv[v], off, 32);
      const int   oi = __shfl_xor(besti[v], off, 32);
      if (ov < bestv[v] || (ov == bestv[v] && oi < besti[v])) {
        bestv[v] = ov; besti[v] = oi;
      }
    }
  }

  // broadcast winning code index of all 16 rows to every lane
  int idxAll[16];
#pragma unroll
  for (int v = 0; v < 8; ++v) {
    idxAll[v]     = __shfl(besti[v], 0, 32);   // rows 0..7  live in lanes 0..15
    idxAll[v + 8] = __shfl(besti[v], 16, 32);  // rows 8..15 live in lanes 16..31
  }

  // ---- histogram for perplexity (L2 atomics, 512 bins) ----
  if (lane < 16) atomicAdd(counts + idxAll[lane], 1u);

  // ---- loss[n] = 1.25 * mean_d (emb[idx,d] - x[n,d])^2 ----
  {
    float* lossOut = out + kLossOff;
    const int d0 = lane * 2;
#pragma unroll 4
    for (int r = 0; r < 16; ++r) {
      const int n  = base + r;
      const int bb = n >> 12;
      const int hw = n & (kHW - 1);
      const float* xr = x + (size_t)bb * kCHW + hw;
      const float* er = emb + (size_t)idxAll[r] * kDim;
      const float e0 = er[d0]     - xr[(size_t)d0 * kHW];
      const float e1 = er[d0 + 1] - xr[(size_t)(d0 + 1) * kHW];
      float s = e0 * e0 + e1 * e1;
#pragma unroll
      for (int off = 16; off >= 1; off >>= 1) s += __shfl_xor(s, off, 32);
      if (lane == 0) lossOut[n] = 1.25f * s * (1.0f / 64.0f);
    }
  }

  // ---- quantized (straight-through == gather), back to NCHW ----
  {
    float* q = out + kQuantOff;
    const int n  = base + nl;
    const int bb = n >> 12;
    const int hw = n & (kHW - 1);
    float* qrow = q + (size_t)bb * kCHW + hw;
    const float* er = emb + (size_t)idxAll[nl] * kDim;
#pragma unroll
    for (int j = 0; j < 32; ++j) {
      const int d = 2 * j + h;   // half-waves interleave d; 16 consecutive hw per store group
      __builtin_nontemporal_store(er[d], qrow + (size_t)d * kHW);
    }
  }

  // ---- one-hot encodings (256 MB > L2 -> NT stores; odd base -> b32) ----
  {
    float* enc = out + kEncOff;
#pragma unroll 2
    for (int r = 0; r < 16; ++r) {
      const int one = idxAll[r];
      float* erow = enc + (size_t)(base + r) * kNumEmb;
#pragma unroll
      for (int c0 = 0; c0 < kNumEmb; c0 += 32) {
        const int c = c0 + lane;
        __builtin_nontemporal_store((c == one) ? 1.0f : 0.0f, erow + c);
      }
    }
  }
}

// ---------------------------------------------------------------------------
// Kernel 3: perplexity = exp(-sum p log(p + 1e-10)), p = counts / N
// ---------------------------------------------------------------------------
__global__ __launch_bounds__(256) void vq_finalize(const unsigned* __restrict__ counts,
                                                   float* __restrict__ out_perp) {
  __shared__ float red[256];
  const int t = threadIdx.x;
  float s = 0.f;
  for (int k = t; k < kNumEmb; k += 256) {
    const float p = (float)counts[k] * (1.0f / (float)kNPix);
    s += p * __logf(p + 1e-10f);
  }
  red[t] = s;
  __syncthreads();
  for (int w = 128; w > 0; w >>= 1) {
    if (t < w) red[t] += red[t + w];
    __syncthreads();
  }
  if (t == 0) out_perp[0] = __expf(-red[0]);
}

// ---------------------------------------------------------------------------
extern "C" void kernel_launch(void* const* d_in, const int* in_sizes, int n_in,
                              void* d_out, int out_size, void* d_ws, size_t ws_size,
                              hipStream_t stream) {
  (void)in_sizes; (void)n_in; (void)out_size; (void)ws_size;
  const float* x   = (const float*)d_in[0];   // [32,64,64,64] NCHW
  const float* emb = (const float*)d_in[1];   // [512,64]
  float* out = (float*)d_out;

  _Float16* embh   = (_Float16*)d_ws;                              // 64 KB
  float*    enorm  = (float*)((char*)d_ws + 65536);                // 2 KB
  unsigned* counts = (unsigned*)((char*)d_ws + 65536 + 2048);      // 2 KB

  vq_prep<<<1, 512, 0, stream>>>(emb, embh, enorm, counts);
  // 8192 tiles of 16 pixels; 8 waves per 256-thread block -> 1024 blocks
  vq_main<<<kNPix / 16 / 8, 256, 0, stream>>>(x, emb, embh, enorm, counts, out);
  vq_finalize<<<1, 256, 0, stream>>>(counts, out + kPerpOff);
}
